// LiquidS4AudioClassifier_22119081575160
// MI455X (gfx1250) — matrix operand order repaired and checked by reading, not compiled, not run on hardware
//
#include <hip/hip_runtime.h>
#include <hip/hip_bf16.h>
#include <math.h>

// ---------------- problem constants (from reference) ----------------
#define HDIM    256          // D_MODEL
#define LSEQ    2048         // SEQ
#define BATCHN  16           // BATCH
#define NSTATE  32           // D_STATE_C
#define NLAYERS 6
#define NMELS   128
#define NCLS    50
#define ROWS    (BATCHN * LSEQ)            // 32768
#define DISCSZ  (NLAYERS * HDIM * NSTATE)  // 49152 per array

typedef __attribute__((ext_vector_type(16))) _Float16 v16h;
typedef __attribute__((ext_vector_type(8)))  _Float16 v8h;
typedef __attribute__((ext_vector_type(8)))  float    v8f;

__device__ __forceinline__ float gelu_tanh(float y) {
    const float c = 0.7978845608028654f;
    float t = tanhf(c * (y + 0.044715f * y * y * y));
    return 0.5f * y * (1.0f + t);
}

// ---------------- 1. ZOH discretization precompute ----------------
// disc layout: 6 arrays of [NL][H][N]: dA_re, dA_im, dB_re, dB_im, C_re, C_im
__global__ void precompute_k(const float* __restrict__ log_dt,
                             const float* __restrict__ A_re, const float* __restrict__ A_im,
                             const float* __restrict__ B_re, const float* __restrict__ B_im,
                             const float* __restrict__ C_re, const float* __restrict__ C_im,
                             float* __restrict__ disc)
{
    int idx = blockIdx.x * blockDim.x + threadIdx.x;
    if (idx >= DISCSZ) return;
    int h  = (idx >> 5) & (HDIM - 1);
    int li = idx >> 13;                      // / (HDIM*NSTATE)
    float dt = expf(log_dt[li * HDIM + h]);
    float Ar = A_re[idx], Ai = A_im[idx];
    float ar = dt * Ar, ai = dt * Ai;
    float e   = expf(ar);
    float dAr = e * cosf(ai);
    float dAi = e * sinf(ai);
    // q = (exp(dtA) - 1) / A   (complex)
    float nr = dAr - 1.0f, ni = dAi;
    float den = Ar * Ar + Ai * Ai;
    float qr = (nr * Ar + ni * Ai) / den;
    float qi = (ni * Ar - nr * Ai) / den;
    float Br = B_re[idx], Bi = B_im[idx];
    disc[0 * DISCSZ + idx] = dAr;
    disc[1 * DISCSZ + idx] = dAi;
    disc[2 * DISCSZ + idx] = Br * qr - Bi * qi;
    disc[3 * DISCSZ + idx] = Br * qi + Bi * qr;
    disc[4 * DISCSZ + idx] = C_re[idx];
    disc[5 * DISCSZ + idx] = C_im[idx];
}

// ---------------- 2. fp32 -> f16 convert (grid-stride) ----------------
__global__ void cvt_f16_k(const float* __restrict__ in, _Float16* __restrict__ out, int n)
{
    int i = blockIdx.x * blockDim.x + threadIdx.x;
    int stride = gridDim.x * blockDim.x;
    for (; i < n; i += stride) out[i] = (_Float16)in[i];
}

// ---------------- 3. Weight repack: f32 [K,N] -> f16 WMMA-B-fragment layout ----------
// Packed layout: tile (tk, tn) covers K rows tk*32..+31, N cols tn*16..+15.
// Within a tile: [lane 0..31][r 0..15] halves, where lane supplies column (lane&15)
// and K rows (lane>>4)*16 + r. Tile block = 512 halves (1024 B), contiguous.
// A wave then fetches its whole B fragment as one 32B load per lane.
__global__ void repack_w_k(const float* __restrict__ W, _Float16* __restrict__ Wp,
                           int K, int N)
{
    int o = blockIdx.x * blockDim.x + threadIdx.x;
    int total = K * N;
    int stride = gridDim.x * blockDim.x;
    int tilesN = N >> 4;
    for (; o < total; o += stride) {
        int r    = o & 15;
        int lane = (o >> 4) & 31;
        int tile = o >> 9;
        int tk   = tile / tilesN;
        int tn   = tile % tilesN;
        int k = tk * 32 + (lane >> 4) * 16 + r;
        int n = tn * 16 + (lane & 15);
        Wp[o] = (_Float16)W[(size_t)k * N + n];
    }
}

// ---------------- 4. WMMA GEMM: C[M,N] = A[M,K](f16 row-major) * Bp(packed f16) ------
// One wave computes a 32x64 tile: 2 A fragments x 4 B fragments = 8 WMMAs / K-step.
// Requires M%32==0, N%64==0, K%32==0. Guards are wave-uniform (EXEC all-1 at WMMA).
__global__ void wmma_gemm_f16_k(const _Float16* __restrict__ A,
                                const _Float16* __restrict__ Bp,
                                float* __restrict__ C,
                                const float* __restrict__ bias,
                                int M, int N, int K)
{
    const int lane   = threadIdx.x & 31;
    const int waveId = blockIdx.x * (blockDim.x >> 5) + (threadIdx.x >> 5);
    const int tilesN = N >> 6;                 // 64-wide wave tiles in N
    const int tm = waveId / tilesN;
    const int tn = waveId % tilesN;
    if (tm * 32 >= M) return;                  // wave-uniform
    const int m0 = tm * 32;
    const int n0 = tn * 64;

    const int mrow = lane & 15;                // A: M row within 16
    const int hi   = lane >> 4;                // lane group 0/1
    const int kA   = hi * 16;                  // A K sub-offset (halves)
    const int ncol = lane & 15;                // C/D column
    const int tilesN16 = N >> 4;               // 16-wide packed B tiles in N

    v8f acc[2][4] = {};

    for (int k0 = 0; k0 < K; k0 += 32) {
        // Two A fragments (rows m0..m0+15, m0+16..m0+31): 2x16B contiguous per lane
        v16h af[2];
        #pragma unroll
        for (int half = 0; half < 2; ++half) {
            const _Float16* arow = A + (size_t)(m0 + half * 16 + mrow) * K + k0;
            v8h alo = *(const v8h*)(arow + kA);
            v8h ahi = *(const v8h*)(arow + kA + 8);
            #pragma unroll
            for (int i = 0; i < 8; ++i) { af[half][i] = alo[i]; af[half][8 + i] = ahi[i]; }
        }

        const size_t tkBase = (size_t)(k0 >> 5) * tilesN16;
        #pragma unroll
        for (int j = 0; j < 4; ++j) {
            // Packed B fragment: one aligned 32B vector load per lane
            const _Float16* bp = Bp + (tkBase + (n0 >> 4) + j) * 512 + lane * 16;
            v16h bf = *(const v16h*)bp;
            acc[0][j] = __builtin_amdgcn_wmma_f32_16x16x32_f16(
                false, af[0], false, bf, (short)0, acc[0][j], false, false);
            acc[1][j] = __builtin_amdgcn_wmma_f32_16x16x32_f16(
                false, af[1], false, bf, (short)0, acc[1][j], false, false);
        }
    }

    const int mb = hi * 8;                     // C/D: M = vgpr + 8*lane[4]
    #pragma unroll
    for (int half = 0; half < 2; ++half) {
        #pragma unroll
        for (int j = 0; j < 4; ++j) {
            #pragma unroll
            for (int r = 0; r < 8; ++r) {
                int m  = m0 + half * 16 + mb + r;
                int nn = n0 + j * 16 + ncol;
                float val = acc[half][j][r];
                if (bias) val += bias[nn];
                C[(size_t)m * N + nn] = val;
            }
        }
    }
}

// ---------------- 5. LayerNorm (one wave per 256-wide row) ----------------
__global__ void layernorm_k(const float* __restrict__ x, float* __restrict__ z,
                            const float* __restrict__ scale, const float* __restrict__ bias,
                            int rows)
{
    int lane = threadIdx.x & 31;
    int row  = blockIdx.x * (blockDim.x >> 5) + (threadIdx.x >> 5);
    if (row >= rows) return;
    const float* xr = x + (size_t)row * HDIM;
    float v[8]; float s = 0.0f;
    #pragma unroll
    for (int i = 0; i < 8; ++i) { v[i] = xr[lane + 32 * i]; s += v[i]; }
    #pragma unroll
    for (int off = 16; off; off >>= 1) s += __shfl_xor(s, off);
    float mu = s * (1.0f / HDIM);
    float var = 0.0f;
    #pragma unroll
    for (int i = 0; i < 8; ++i) { float d = v[i] - mu; var += d * d; }
    #pragma unroll
    for (int off = 16; off; off >>= 1) var += __shfl_xor(var, off);
    float rs = rsqrtf(var * (1.0f / HDIM) + 1e-5f);
    float* zr = z + (size_t)row * HDIM;
    #pragma unroll
    for (int i = 0; i < 8; ++i) {
        int c = lane + 32 * i;
        zr[c] = (v[i] - mu) * rs * scale[c] + bias[c];
    }
}

// ---------------- 6. SSM scan: one wave per (b,h), lane = complex state n ----------------
// Fuses D-skip + tanh-GELU + f16 conversion on the output (ready for WMMA GEMM).
__global__ void ssm_scan_k(const float* __restrict__ z, _Float16* __restrict__ y16,
                           const float* __restrict__ disc, const float* __restrict__ Dvec,
                           int layer)
{
    int lane = threadIdx.x & 31;
    int bh   = blockIdx.x * (blockDim.x >> 5) + (threadIdx.x >> 5);   // 0..B*H-1
    int b = bh >> 8;
    int h = bh & (HDIM - 1);
    int base = (layer * HDIM + h) * NSTATE + lane;
    float dAr = disc[0 * DISCSZ + base];
    float dAi = disc[1 * DISCSZ + base];
    float dBr = disc[2 * DISCSZ + base];
    float dBi = disc[3 * DISCSZ + base];
    float Cr  = disc[4 * DISCSZ + base];
    float Ci  = disc[5 * DISCSZ + base];
    float Dh  = Dvec[layer * HDIM + h];

    const float* zp = z   + ((size_t)b * LSEQ) * HDIM + h;
    _Float16*    yp = y16 + ((size_t)b * LSEQ) * HDIM + h;

    float sr = 0.0f, si = 0.0f;
    for (int l = 0; l < LSEQ; ++l) {
        if ((l & 63) == 0 && l + 128 < LSEQ)
            __builtin_prefetch(zp + (size_t)(l + 128) * HDIM, 0, 1);  // global_prefetch_b8
        float u = zp[(size_t)l * HDIM];
        float nsr = dAr * sr - dAi * si + dBr * u;
        float nsi = dAr * si + dAi * sr + dBi * u;
        sr = nsr; si = nsi;
        float p = Cr * sr - Ci * si;
        #pragma unroll
        for (int off = 16; off; off >>= 1) p += __shfl_xor(p, off);
        if (lane == 0) {
            float yv = 2.0f * p + Dh * u;
            yp[(size_t)l * HDIM] = (_Float16)gelu_tanh(yv);
        }
    }
}

// ---------------- 7. GLU + residual: x += a * sigmoid(g) ----------------
__global__ void glu_res_k(const float* __restrict__ v, const float* __restrict__ b_out,
                          float* __restrict__ x, int layer)
{
    int i = blockIdx.x * blockDim.x + threadIdx.x;
    int stride = gridDim.x * blockDim.x;
    const float* bo = b_out + (size_t)layer * (2 * HDIM);
    const int total = ROWS * HDIM;
    for (; i < total; i += stride) {
        int hc = i & (HDIM - 1);
        size_t row = (size_t)(i >> 8);
        float a = v[row * (2 * HDIM) + hc]        + bo[hc];
        float g = v[row * (2 * HDIM) + HDIM + hc] + bo[HDIM + hc];
        x[i] += a * (1.0f / (1.0f + expf(-g)));
    }
}

// ---------------- 8. Mean pool over time ----------------
__global__ void pool_k(const float* __restrict__ z, float* __restrict__ pooled)
{
    int bh = blockIdx.x * blockDim.x + threadIdx.x;   // B*H = 4096
    int b = bh >> 8, h = bh & (HDIM - 1);
    float s = 0.0f;
    const float* zp = z + ((size_t)b * LSEQ) * HDIM + h;
    for (int l = 0; l < LSEQ; ++l) s += zp[(size_t)l * HDIM];
    pooled[bh] = s * (1.0f / LSEQ);
}

// ---------------- 9. Classifier (tiny; scalar) ----------------
__global__ void cls_k(const float* __restrict__ pooled, const float* __restrict__ Wc,
                      const float* __restrict__ bc, float* __restrict__ out)
{
    int i = blockIdx.x * blockDim.x + threadIdx.x;
    if (i >= BATCHN * NCLS) return;
    int b = i / NCLS, c = i % NCLS;
    float s = bc[c];
    const float* pb = pooled + b * HDIM;
    for (int h = 0; h < HDIM; ++h) s += pb[h] * Wc[h * NCLS + c];
    out[i] = s;
}

// ---------------- launch ----------------
extern "C" void kernel_launch(void* const* d_in, const int* in_sizes, int n_in,
                              void* d_out, int out_size, void* d_ws, size_t ws_size,
                              hipStream_t stream)
{
    const float* x      = (const float*)d_in[0];
    const float* W_in   = (const float*)d_in[1];
    const float* b_in   = (const float*)d_in[2];
    const float* log_dt = (const float*)d_in[3];
    const float* A_re   = (const float*)d_in[4];
    const float* A_im   = (const float*)d_in[5];
    const float* B_re   = (const float*)d_in[6];
    const float* B_im   = (const float*)d_in[7];
    const float* C_re   = (const float*)d_in[8];
    const float* C_im   = (const float*)d_in[9];
    const float* Dv     = (const float*)d_in[10];
    const float* W_out  = (const float*)d_in[11];
    const float* b_out  = (const float*)d_in[12];
    const float* ln_s   = (const float*)d_in[13];
    const float* ln_b   = (const float*)d_in[14];
    const float* lnf_s  = (const float*)d_in[15];
    const float* lnf_b  = (const float*)d_in[16];
    const float* W_cls  = (const float*)d_in[17];
    const float* b_cls  = (const float*)d_in[18];
    float* out = (float*)d_out;

    // workspace carve-out (all 256B aligned)
    char* ws = (char*)d_ws;
    size_t off = 0;
    auto carve = [&](size_t bytes) -> char* {
        char* p = ws + off;
        off += (bytes + 255) & ~(size_t)255;
        return p;
    };
    float*    xbuf   = (float*)   carve((size_t)ROWS * HDIM * 4);       // activations
    float*    zbuf   = (float*)   carve((size_t)ROWS * HDIM * 4);       // LN output
    float*    vbuf   = (float*)   carve((size_t)ROWS * 2 * HDIM * 4);   // W_out GEMM out
    _Float16* y16    = (_Float16*)carve((size_t)ROWS * HDIM * 2);       // f16 GEMM input (also x16 staging)
    _Float16* w16p   = (_Float16*)carve((size_t)HDIM * 2 * HDIM * 2);   // packed f16 weights
    float*    disc   = (float*)   carve((size_t)6 * DISCSZ * 4);        // dA,dB,C
    float*    pooled = (float*)   carve((size_t)BATCHN * HDIM * 4);

    // 1. discretization precompute
    precompute_k<<<(DISCSZ + 255) / 256, 256, 0, stream>>>(
        log_dt, A_re, A_im, B_re, B_im, C_re, C_im, disc);

    // 2. input projection: x[32768,128] @ W_in[128,256] + b_in
    cvt_f16_k<<<2048, 256, 0, stream>>>(x, y16, ROWS * NMELS);
    repack_w_k<<<128, 256, 0, stream>>>(W_in, w16p, NMELS, HDIM);
    {
        int waves = (ROWS / 32) * (HDIM / 64);          // 4096
        wmma_gemm_f16_k<<<waves / 8, 256, 0, stream>>>(y16, w16p, xbuf, b_in,
                                                       ROWS, HDIM, NMELS);
    }

    // 3. layers
    for (int i = 0; i < NLAYERS; ++i) {
        layernorm_k<<<ROWS / 8, 256, 0, stream>>>(xbuf, zbuf,
                                                  ln_s + i * HDIM, ln_b + i * HDIM, ROWS);
        ssm_scan_k<<<(BATCHN * HDIM) / 8, 256, 0, stream>>>(zbuf, y16, disc, Dv, i);
        repack_w_k<<<512, 256, 0, stream>>>(W_out + (size_t)i * HDIM * 2 * HDIM,
                                            w16p, HDIM, 2 * HDIM);
        {
            int waves = (ROWS / 32) * ((2 * HDIM) / 64); // 8192
            wmma_gemm_f16_k<<<waves / 8, 256, 0, stream>>>(y16, w16p, vbuf, nullptr,
                                                           ROWS, 2 * HDIM, HDIM);
        }
        glu_res_k<<<4096, 256, 0, stream>>>(vbuf, b_out, xbuf, i);
    }

    // 4. final LN + pool + classifier
    layernorm_k<<<ROWS / 8, 256, 0, stream>>>(xbuf, zbuf, lnf_s, lnf_b, ROWS);
    pool_k<<<(BATCHN * HDIM) / 256, 256, 0, stream>>>(zbuf, pooled);
    cls_k<<<4, 256, 0, stream>>>(pooled, W_cls, b_cls, out);
}
